// vgg16_Dropout_4449586118759
// MI455X (gfx1250) — compile-verified, matrix operand order checked by reading, not run on hardware
//
#include <hip/hip_runtime.h>
#include <hip/hip_bf16.h>
#include <math.h>

// ---------------------------------------------------------------------------
// CDNA5 (gfx1250) spherical-CNN forward, v2.
// - Activations stored bf16; gather-GEMM uses v_wmma_f32_16x16x32_bf16.
// - K restructured as neighbor-j (unrolled x7) x channel blocks: no division,
//   no dynamic register indexing in the hot loop.
// - A fragment: two contiguous 16-byte bf16 loads per K-step.
// - W pre-packed per layer into fragment-major bf16: one contiguous 32-byte
//   load per lane per K-step.
// Workspace: h fp32 (21MB) | act0 bf16 (10.5MB) | act1 bf16 (10.5MB) |
//            packedW bf16 (14.7MB) | stats (~20KB)  => ~56.7 MB.
// ---------------------------------------------------------------------------

typedef __attribute__((ext_vector_type(16))) __bf16 bfvec16;
typedef __attribute__((ext_vector_type(8)))  __bf16 bfvec8;
typedef __attribute__((ext_vector_type(8)))  float  fvec8;

__device__ __forceinline__ __bf16 f2bf(float f) {
  unsigned u = __builtin_bit_cast(unsigned, f);
  unsigned r = (u + 0x7FFFu + ((u >> 16) & 1u)) >> 16;   // round-to-nearest-even
  unsigned short s = (unsigned short)r;
  return __builtin_bit_cast(__bf16, s);
}
__device__ __forceinline__ float bf2f(__bf16 b) {
  unsigned short s = __builtin_bit_cast(unsigned short, b);
  unsigned u = ((unsigned)s) << 16;
  return __builtin_bit_cast(float, u);
}

// --------------------------- fused gather-GEMM -----------------------------
// 256 threads = 8 waves; wave w computes the 16x16 tile (tm = bx*8+w, tn=by).
// act: bf16 [Nrows, cip] row-major (cip % 32 == 0).
// pw : packed bf16 W fragments: [(kb*2+half)*co + n] * 16 contiguous elems.
__global__ void gemm_gather_bf16(const __bf16* __restrict__ act,
                                 const int*    __restrict__ no,
                                 const __bf16* __restrict__ pw,
                                 const float*  __restrict__ bias,
                                 float* __restrict__ h,
                                 float* __restrict__ csum,
                                 float* __restrict__ csq,
                                 int Nrows, int cip, int co) {
  const int lane  = threadIdx.x & 31;
  const int wave  = threadIdx.x >> 5;
  const int mlane = lane & 15;
  const int half  = lane >> 4;
  const int tm = blockIdx.x * 8 + wave;
  const int tn = blockIdx.y;
  if (tm * 16 >= Nrows) return;               // uniform per wave

  const int row = tm * 16 + mlane;
  const int col = tn * 16 + mlane;
  const bool rowok = row < Nrows;

  int nb[7];
#pragma unroll
  for (int j = 0; j < 7; ++j)
    nb[j] = rowok ? no[7 * row + j] * cip : 0;

  fvec8 acc = {};
  int kb = 0;
#pragma unroll
  for (int j = 0; j < 7; ++j) {
    const __bf16* abase = act + nb[j] + 8 * half;
    for (int c0 = 0; c0 < cip; c0 += 32, ++kb) {
      // A fragment: slots 0..7 <- channels c0+8*half+[0..7],
      //             slots 8..15 <- channels c0+16+8*half+[0..7]
      bfvec8 alo = *(const bfvec8*)(abase + c0);
      bfvec8 ahi = *(const bfvec8*)(abase + c0 + 16);
      bfvec16 a = __builtin_shufflevector(alo, ahi,
          0, 1, 2, 3, 4, 5, 6, 7, 8, 9, 10, 11, 12, 13, 14, 15);
      // B fragment: one contiguous 32-byte load from packed W.
      bfvec16 b = *(const bfvec16*)(pw +
          ((((size_t)(kb * 2 + half)) * co + col) << 4));
      acc = __builtin_amdgcn_wmma_f32_16x16x32_bf16(
          false, a, false, b, (short)0, acc, false, false);
    }
  }

  // C/D layout: VGPR v holds M = v + 8*half, N = mlane.
  const float bcol = bias[col];
  float s = 0.f, q = 0.f;
#pragma unroll
  for (int v = 0; v < 8; ++v) {
    const int r = tm * 16 + v + (half ? 8 : 0);
    const float val = acc[v] + bcol;
    if (r < Nrows) {
      h[(size_t)r * co + col] = val;
      s += val;
      q += val * val;
    }
  }
  s += __shfl_xor(s, 16, 32);
  q += __shfl_xor(q, 16, 32);
  if (half == 0) {
    atomicAdd(&csum[col], s);
    atomicAdd(&csq[col],  q);
  }
}

// --------------------------- W fragment packing ----------------------------
// Input W: fp32 [7*ci, co]. Output: bf16 fragments, zero-padded to cip.
__global__ void pack_w(const float* __restrict__ W, __bf16* __restrict__ pw,
                       int ci, int cip, int co) {
  const int idx = blockIdx.x * blockDim.x + threadIdx.x;
  const int total = 7 * cip * co;
  if (idx >= total) return;
  const int kp = idx / co;
  const int n  = idx - kp * co;
  const int j  = kp / cip;
  const int c  = kp - j * cip;
  const float v = (c < ci) ? W[(size_t)(j * ci + c) * co + n] : 0.f;
  const int kb = kp >> 5, r = kp & 31;
  const int half = (r >> 3) & 1, hi = (r >> 4) & 1, q = r & 7;
  const size_t o = ((((size_t)(kb * 2 + half)) * co + n) << 4) + 8 * hi + q;
  pw[o] = f2bf(v);
}

// --------------------------- input convert/pad -----------------------------
// x fp32 [N,3] -> bf16 [N,32] zero-padded.
__global__ void convert_pad_x(const float* __restrict__ x,
                              __bf16* __restrict__ out, int N) {
  const int i = blockIdx.x * blockDim.x + threadIdx.x;
  if (i < N * 32) {
    const int v = i >> 5, c = i & 31;
    out[i] = f2bf(c < 3 ? x[v * 3 + c] : 0.f);
  }
}

// ------------------------------ BN helpers ---------------------------------
__global__ void zero_stats(float* a, float* b, int n) {
  const int i = blockIdx.x * blockDim.x + threadIdx.x;
  if (i < n) { a[i] = 0.f; b[i] = 0.f; }
}

__global__ void finalize_bn(const float* __restrict__ csum,
                            const float* __restrict__ csq,
                            const float* __restrict__ gamma,
                            const float* __restrict__ beta,
                            float* __restrict__ scale,
                            float* __restrict__ shift,
                            int Nrows, int co) {
  const int c = blockIdx.x * blockDim.x + threadIdx.x;
  if (c < co) {
    const float inv = 1.f / (float)Nrows;
    const float mu  = csum[c] * inv;
    const float var = csq[c] * inv - mu * mu;
    const float sc  = gamma[c] * rsqrtf(var + 1e-5f);
    scale[c] = sc;
    shift[c] = beta[c] - mu * sc;
  }
}

// normalize + leaky-ReLU; write next activation as bf16.
__global__ void bn_act(const float* __restrict__ h,
                       __bf16* __restrict__ out,
                       const float* __restrict__ scale,
                       const float* __restrict__ shift,
                       int total, int co) {
  const int i = blockIdx.x * blockDim.x + threadIdx.x;
  if (i < total) {
    const int c = i % co;
    float v = h[i] * scale[c] + shift[c];
    v = (v > 0.f) ? v : 0.1f * v;
    out[i] = f2bf(v);
  }
}

// ------------------------------- pooling -----------------------------------
__global__ void pool7(const __bf16* __restrict__ act,
                      const int*    __restrict__ no_fine,
                      __bf16* __restrict__ out, int ncoarse, int co) {
  const int i = blockIdx.x * blockDim.x + threadIdx.x;
  if (i < ncoarse * co) {
    const int vtx = i / co;
    const int c   = i - vtx * co;
    float s = 0.f;
#pragma unroll
    for (int j = 0; j < 7; ++j)
      s += bf2f(act[(size_t)no_fine[7 * vtx + j] * co + c]);
    out[i] = f2bf(s * (1.f / 7.f));
  }
}

// --------------------------------- head ------------------------------------
__global__ void colmean(const __bf16* __restrict__ act,
                        float* __restrict__ mean, int Nrows, int co) {
  const int c = blockIdx.x * blockDim.x + threadIdx.x;
  if (c < co) {
    float s = 0.f;
    for (int r = 0; r < Nrows; ++r) s += bf2f(act[(size_t)r * co + c]);
    mean[c] = s / (float)Nrows;
  }
}

__global__ void final_dot(const float* __restrict__ mean,
                          const float* __restrict__ fcW,
                          const float* __restrict__ fcb,
                          float* __restrict__ out, int co) {
  __shared__ float red[256];
  float p = 0.f;
  for (int c = threadIdx.x; c < co; c += 256) p += mean[c] * fcW[c];
  red[threadIdx.x] = p;
  __syncthreads();
  for (int s = 128; s > 0; s >>= 1) {
    if (threadIdx.x < s) red[threadIdx.x] += red[threadIdx.x + s];
    __syncthreads();
  }
  if (threadIdx.x == 0) out[0] = red[0] + fcb[0];
}

// ------------------------------ orchestration ------------------------------
extern "C" void kernel_launch(void* const* d_in, const int* in_sizes, int n_in,
                              void* d_out, int out_size, void* d_ws, size_t ws_size,
                              hipStream_t stream) {
  (void)in_sizes; (void)n_in; (void)out_size; (void)ws_size;

  // Input order: x, no0..no5, W[0..12], b[0..12], gamma[0..12], beta[0..12],
  //              fc_W, fc_b  (setup_inputs dict order, params flattened).
  const float* x = (const float*)d_in[0];
  const int* no[6];
  for (int i = 0; i < 6; ++i) no[i] = (const int*)d_in[1 + i];
  const float *W[13], *bias[13], *gamma[13], *beta[13];
  for (int k = 0; k < 13; ++k) {
    W[k]     = (const float*)d_in[7 + k];
    bias[k]  = (const float*)d_in[20 + k];
    gamma[k] = (const float*)d_in[33 + k];
    beta[k]  = (const float*)d_in[46 + k];
  }
  const float* fcW = (const float*)d_in[59];
  const float* fcb = (const float*)d_in[60];

  static const int NS_[6]  = {163842, 40962, 10242, 2562, 642, 162};
  static const int ci_[13] = {3, 32, 32, 32, 64, 64, 128, 128, 256, 256, 512, 512, 1024};
  static const int co_[13] = {32, 32, 32, 64, 64, 128, 128, 256, 256, 512, 512, 1024, 1024};
  static const int lv_[13] = {0, 0, 0, 1, 1, 2, 2, 3, 3, 4, 4, 5, 5};

  char* wsb = (char*)d_ws;
  const size_t ACT_ELEMS = (size_t)163842 * 32;          // 5,242,944
  float*  hbuf  = (float*)(wsb);                         // 20,971,776 B
  __bf16* actb[2];
  actb[0] = (__bf16*)(wsb + 20971776);                   // 10,485,888 B
  actb[1] = (__bf16*)(wsb + 31457664);                   // 10,485,888 B
  __bf16* pw    = (__bf16*)(wsb + 41943552);             // 14,680,064 B
  float*  csum  = (float*)(wsb + 56623616);
  float*  csq   = csum + 1024;
  float*  scale = csq + 1024;
  float*  shift = scale + 1024;
  float*  meanv = shift + 1024;
  (void)ACT_ELEMS;

  // Input: fp32 [163842,3] -> bf16 [163842,32] zero-padded.
  {
    const int total = 163842 * 32;
    convert_pad_x<<<(total + 255) / 256, 256, 0, stream>>>(x, actb[0], 163842);
  }
  const __bf16* cur = actb[0];
  int flip = 1;

  for (int layer = 0; layer < 13; ++layer) {
    const int l = lv_[layer];

    // Pool from the finer level before the first conv of levels 1..5.
    if (layer == 3 || layer == 5 || layer == 7 || layer == 9 || layer == 11) {
      const int ncoarse = NS_[l];
      const int cch = ci_[layer];
      __bf16* dst = actb[flip];
      const int total = ncoarse * cch;
      pool7<<<(total + 255) / 256, 256, 0, stream>>>(cur, no[l - 1], dst, ncoarse, cch);
      cur = dst;
      flip ^= 1;
    }

    const int N = NS_[l], ci = ci_[layer], co = co_[layer];
    const int cip = (ci < 32) ? 32 : ci;                  // padded channels

    {
      const int total = 7 * cip * co;
      pack_w<<<(total + 255) / 256, 256, 0, stream>>>(W[layer], pw, ci, cip, co);
    }
    zero_stats<<<4, 256, 0, stream>>>(csum, csq, 1024);

    const int mtiles = (N + 15) / 16;
    dim3 grid((mtiles + 7) / 8, co / 16);
    gemm_gather_bf16<<<grid, 256, 0, stream>>>(cur, no[l], pw, bias[layer],
                                               hbuf, csum, csq, N, cip, co);

    finalize_bn<<<(co + 255) / 256, 256, 0, stream>>>(csum, csq, gamma[layer],
                                                      beta[layer], scale, shift, N, co);

    const int total = N * co;
    __bf16* nxt = actb[flip];
    bn_act<<<(total + 255) / 256, 256, 0, stream>>>(hbuf, nxt, scale, shift, total, co);

    cur = nxt;
    flip ^= 1;
  }

  // Head: mean over 162 vertices -> dot with fc_W -> scalar.
  colmean<<<4, 256, 0, stream>>>(cur, meanv, 162, 1024);
  final_dot<<<1, 256, 0, stream>>>(meanv, fcW, fcb, (float*)d_out, 1024);
}
